// VectorQuantizer_7129645711678
// MI455X (gfx1250) — matrix operand order repaired and checked by reading, not compiled, not run on hardware
//
#include <hip/hip_runtime.h>
#include <hip/hip_bf16.h>
#include <stdint.h>

// Problem constants (match reference)
#define K_NUM   8192
#define D_DIM   512
#define BATCH_N 16384
#define MT      128    // batch rows per block in argmin kernel
#define NT      16     // codebook rows per WMMA tile
#define NTILES  (K_NUM / NT)   // 512

#define AS1 __attribute__((address_space(1)))
#define AS3 __attribute__((address_space(3)))

#if defined(__has_builtin)
#  if __has_builtin(__builtin_amdgcn_global_load_async_to_lds_b128)
#    define VQ_ASYNC 1
#  endif
#endif
#ifndef VQ_ASYNC
#  define VQ_ASYNC 0
#endif

typedef __bf16 bf16_t;
typedef __bf16 v16bf __attribute__((ext_vector_type(16)));
typedef float  v8f   __attribute__((ext_vector_type(8)));
typedef __bf16 v4bf  __attribute__((ext_vector_type(4)));
typedef int    v4i   __attribute__((ext_vector_type(4)));

union Frag { uint4 u[2]; v16bf v; };

// 16-byte global->LDS copy; async-to-LDS (ASYNCcnt) when available.
__device__ __forceinline__ void cp16(const void* g, void* l) {
#if VQ_ASYNC
  // Generic LDS address low 32 bits == LDS byte offset (ISA 10.2 aperture map),
  // so inttoptr to AS3 yields the correct ds address. Global generic == AS1.
  __builtin_amdgcn_global_load_async_to_lds_b128(
      (AS1 v4i*)(uintptr_t)g,
      (AS3 v4i*)(uint32_t)(uintptr_t)l,
      0, 0);
#else
  *(uint4*)l = *(const uint4*)g;
#endif
}

__device__ __forceinline__ void wait_async0() {
#if VQ_ASYNC
#  if __has_builtin(__builtin_amdgcn_s_wait_asynccnt)
  __builtin_amdgcn_s_wait_asynccnt(0);
#  else
  asm volatile("s_wait_asynccnt 0" ::: "memory");
#  endif
#endif
}

// ---------------------------------------------------------------------------
// Kernel 1: convert codebook fp32 -> bf16, compute per-row squared norms.
// ---------------------------------------------------------------------------
__global__ __launch_bounds__(128) void vq_prep(const float* __restrict__ W,
                                               bf16_t* __restrict__ Wb,
                                               float* __restrict__ norms) {
  const int k = blockIdx.x;
  const int t = threadIdx.x;
  const float4* row = (const float4*)(W + (size_t)k * D_DIM);
  float4 f = row[t];
  float ss = f.x * f.x + f.y * f.y + f.z * f.z + f.w * f.w;

  v4bf o;
  o[0] = (__bf16)f.x; o[1] = (__bf16)f.y; o[2] = (__bf16)f.z; o[3] = (__bf16)f.w;
  ((v4bf*)(Wb + (size_t)k * D_DIM))[t] = o;

  for (int off = 16; off; off >>= 1) ss += __shfl_down(ss, off, 32);
  __shared__ float red[4];
  if ((t & 31) == 0) red[t >> 5] = ss;
  __syncthreads();
  if (t == 0) norms[k] = red[0] + red[1] + red[2] + red[3];
}

// ---------------------------------------------------------------------------
// Kernel 2: loss = 0.25 * sum(W^2). (diff^2 term added by vq_gather.)
// ---------------------------------------------------------------------------
__global__ __launch_bounds__(256) void vq_loss(const float* __restrict__ norms,
                                               float* __restrict__ out) {
  float s = 0.f;
  for (int i = threadIdx.x; i < K_NUM; i += 256) s += norms[i];
  for (int off = 16; off; off >>= 1) s += __shfl_down(s, off, 32);
  __shared__ float red[8];
  if ((threadIdx.x & 31) == 0) red[threadIdx.x >> 5] = s;
  __syncthreads();
  if (threadIdx.x == 0) {
    float t = 0.f;
    #pragma unroll
    for (int i = 0; i < 8; ++i) t += red[i];
    out[0] = 0.25f * t;
  }
}

// ---------------------------------------------------------------------------
// Kernel 3: fused GEMM + argmin.  score(b,k) = ||W_k||^2 - 2 * W[x_b] . W_k
// 128 threads = 4 waves; wave w owns 32 batch rows (two 16-row M-subtiles
// sharing each B fragment -> 6 ds_load_b128 per 2 wmma).  B tile is
// double-buffered and staged with GLOBAL_LOAD_ASYNC_TO_LDS_B128 (ASYNCcnt),
// overlapping the copy with WMMA compute; one barrier per tile.
// LDS: 128KB (A) + 2x16KB (B) = 160KB -> 2 blocks per 320KB WGP.
// ---------------------------------------------------------------------------
__global__ __launch_bounds__(128) void vq_argmin(const int* __restrict__ x,
                                                 const bf16_t* __restrict__ Wb,
                                                 const float* __restrict__ norms,
                                                 int* __restrict__ assign) {
  __shared__ __align__(16) uint16_t sA[MT * D_DIM];      // 128 KB gathered A rows
  __shared__ __align__(16) uint16_t sB[2 * NT * D_DIM];  // 2 x 16 KB B ping/pong

  const int tid     = threadIdx.x;
  const int wave    = tid >> 5;
  const int lane    = tid & 31;
  const int l16     = lane & 15;
  const int hi      = lane >> 4;       // K-half select per 16-bit layouts
  const int rowBase = blockIdx.x * MT;

  // --- stage A: MT gathered rows, 64 x 16B chunks per row (async) ---
  for (int i = tid; i < MT * 64; i += 128) {
    const int r = i >> 6;
    const int c = i & 63;
    const int src = x[rowBase + r];
    cp16(Wb + (size_t)src * D_DIM + (size_t)c * 8,
         sA + (size_t)r * D_DIM + (size_t)c * 8);
  }
  // --- stage B tile 0 into buffer 0 ---
  #pragma unroll
  for (int i = 0; i < 8; ++i) {
    const int c = tid + i * 128;                 // 1024 x 16B chunks
    cp16(Wb + (size_t)c * 8, sB + (size_t)c * 8);
  }
  wait_async0();
  __syncthreads();

  float bv0[8], bv1[8];
  int   bi0[8], bi1[8];
  #pragma unroll
  for (int i = 0; i < 8; ++i) {
    bv0[i] = 3.4e38f; bi0[i] = 0x7fffffff;
    bv1[i] = 3.4e38f; bi1[i] = 0x7fffffff;
  }

  // A fragment rows for the wave's two M-subtiles (row stride = 64 uint4).
  const uint4* sArow0 = (const uint4*)(sA + (size_t)(wave * 32 + l16) * D_DIM);
  const uint4* sArow1 = sArow0 + 16 * (D_DIM / 8);

  for (int nt = 0; nt < NTILES; ++nt) {
    const int cur = nt & 1;

    // Stage next tile into the other buffer (overlaps with compute below).
    if (nt + 1 < NTILES) {
      const bf16_t* gB = Wb + (size_t)(nt + 1) * NT * D_DIM;
      uint16_t* lB = sB + (size_t)(1 - cur) * NT * D_DIM;
      #pragma unroll
      for (int i = 0; i < 8; ++i) {
        const int c = tid + i * 128;
        cp16(gB + (size_t)c * 8, lB + (size_t)c * 8);
      }
    }
    if (nt + 2 < NTILES)   // pull tile nt+2 toward near caches
      __builtin_prefetch(Wb + (size_t)(nt + 2) * NT * D_DIM + (size_t)tid * 128, 0, 3);

    const uint4* sBrow = (const uint4*)(sB + (size_t)(cur * NT + l16) * D_DIM);

    v8f c0 = {0.f, 0.f, 0.f, 0.f, 0.f, 0.f, 0.f, 0.f};
    v8f c1 = {0.f, 0.f, 0.f, 0.f, 0.f, 0.f, 0.f, 0.f};
    #pragma unroll
    for (int d = 0; d < 16; ++d) {
      Frag a0, a1, b;
      // A (16x32 bf16): lane lo: K0-7 (v0-3) + K16-23 (v4-7); lane hi: +8
      a0.u[0] = sArow0[d * 4 + hi];
      a0.u[1] = sArow0[d * 4 + 2 + hi];
      a1.u[0] = sArow1[d * 4 + hi];
      a1.u[1] = sArow1[d * 4 + 2 + hi];
      // B (32x16 bf16): lane lo: K0-7 + K16-23; lane hi: +8 (matches A)
      b.u[0] = sBrow[d * 4 + hi];
      b.u[1] = sBrow[d * 4 + 2 + hi];
      c0 = __builtin_amdgcn_wmma_f32_16x16x32_bf16(false, a0.v, false, b.v,
                                                   (short)0, c0, false, false);
      c1 = __builtin_amdgcn_wmma_f32_16x16x32_bf16(false, a1.v, false, b.v,
                                                   (short)0, c1, false, false);
    }

    // Epilogue: c0[i] -> row (32w + i + 8hi), c1[i] -> row (32w+16 + i + 8hi),
    // column nt*16 + l16.
    const int nCol = nt * NT + l16;
    const float nrm = norms[nCol];
    #pragma unroll
    for (int i = 0; i < 8; ++i) {
      const float s0 = nrm - 2.0f * c0[i];
      if (s0 < bv0[i]) { bv0[i] = s0; bi0[i] = nCol; }
      else if (s0 == bv0[i] && nCol < bi0[i]) { bi0[i] = nCol; }
      const float s1 = nrm - 2.0f * c1[i];
      if (s1 < bv1[i]) { bv1[i] = s1; bi1[i] = nCol; }
      else if (s1 == bv1[i] && nCol < bi1[i]) { bi1[i] = nCol; }
    }

    wait_async0();     // our async writes to the other buffer are in LDS
    __syncthreads();   // everyone done reading cur / writing next
  }

  // Cross-lane (min, argmin) reduction within each 16-lane half.
  #pragma unroll
  for (int i = 0; i < 8; ++i) {
    for (int off = 8; off; off >>= 1) {
      float ov = __shfl_xor(bv0[i], off, 16);
      int   oi = __shfl_xor(bi0[i], off, 16);
      if (ov < bv0[i] || (ov == bv0[i] && oi < bi0[i])) { bv0[i] = ov; bi0[i] = oi; }
      ov = __shfl_xor(bv1[i], off, 16);
      oi = __shfl_xor(bi1[i], off, 16);
      if (ov < bv1[i] || (ov == bv1[i] && oi < bi1[i])) { bv1[i] = ov; bi1[i] = oi; }
    }
    if (l16 == 0) {
      assign[rowBase + wave * 32 + hi * 8 + i]      = bi0[i];
      assign[rowBase + wave * 32 + 16 + hi * 8 + i] = bi1[i];
    }
  }
}

// ---------------------------------------------------------------------------
// Kernel 4: gathers + diff + loss correction, all in fp32 (exact).
// out layout: [0]=loss, [1 .. B*D]=quantized, [1+B*D .. 1+2*B*D]=diff
// ---------------------------------------------------------------------------
__global__ __launch_bounds__(128) void vq_gather(const int* __restrict__ x,
                                                 const float* __restrict__ W,
                                                 const int* __restrict__ assign,
                                                 float* __restrict__ out) {
  const int row = blockIdx.x;
  const int t = threadIdx.x;
  const int a  = assign[row];
  const int xi = x[row];
  const float4 q = ((const float4*)(W + (size_t)a  * D_DIM))[t];
  const float4 e = ((const float4*)(W + (size_t)xi * D_DIM))[t];
  const float4 d = {q.x - e.x, q.y - e.y, q.z - e.z, q.w - e.w};

  float* outq = out + 1 + (size_t)row * D_DIM + (size_t)t * 4;
  float* outd = out + 1 + (size_t)BATCH_N * D_DIM + (size_t)row * D_DIM + (size_t)t * 4;
  outq[0] = q.x; outq[1] = q.y; outq[2] = q.z; outq[3] = q.w;
  outd[0] = d.x; outd[1] = d.y; outd[2] = d.z; outd[3] = d.w;

  float ss = d.x * d.x + d.y * d.y + d.z * d.z + d.w * d.w;
  for (int off = 16; off; off >>= 1) ss += __shfl_down(ss, off, 32);
  __shared__ float red[4];
  if ((t & 31) == 0) red[t >> 5] = ss;
  __syncthreads();
  if (t == 0) {
    const float s = red[0] + red[1] + red[2] + red[3];
    if (s != 0.f) atomicAdd(out, s * (1.0f / BATCH_N));  // zero in practice -> deterministic
  }
}

// ---------------------------------------------------------------------------
extern "C" void kernel_launch(void* const* d_in, const int* in_sizes, int n_in,
                              void* d_out, int out_size, void* d_ws, size_t ws_size,
                              hipStream_t stream) {
  (void)in_sizes; (void)n_in; (void)out_size; (void)ws_size;

  const int*   x = (const int*)d_in[0];
  const float* W = (const float*)d_in[1];
  float* out = (float*)d_out;

  uint8_t* ws    = (uint8_t*)d_ws;
  bf16_t*  Wb    = (bf16_t*)ws;                                       // 8 MB
  float*   norms = (float*)(ws + (size_t)K_NUM * D_DIM * 2);          // 32 KB
  int*     assign = (int*)(ws + (size_t)K_NUM * D_DIM * 2 + (size_t)K_NUM * 4); // 64 KB

  vq_prep  <<<K_NUM,        128, 0, stream>>>(W, Wb, norms);
  vq_loss  <<<1,            256, 0, stream>>>(norms, out);
  vq_argmin<<<BATCH_N / MT, 128, 0, stream>>>(x, Wb, norms, assign);
  vq_gather<<<BATCH_N,      128, 0, stream>>>(x, W, assign, out);
}
